// Encoder_5171140624511
// MI455X (gfx1250) — matrix operand order repaired and checked by reading, not compiled, not run on hardware
//
#include <hip/hip_runtime.h>
#include <cstddef>
#include <cstdint>

// Problem constants (match reference)
#define B_     256
#define T_     1024
#define DIN_   256
#define H_     512
#define FOURH_ 2048
#define L_     128
#define K1_    768    // DIN + H   (layer-1 K: [x_t ; h1])
#define K2_    1024   // H + H     (layer-2 K: [h1 ; h2])
#define NWG_   64     // persistent workgroups: 4 M-tiles(64) x 16 N-tiles(32 hidden cols)

typedef __attribute__((ext_vector_type(16))) _Float16 v16h;
typedef __attribute__((ext_vector_type(8)))  _Float16 v8h;
typedef __attribute__((ext_vector_type(8)))  float    v8f;

// ---------------------------------------------------------------------------
// Prep kernel 1: repack fp32 weight [K,N] (row-major) into WMMA-B-friendly
// f16 layout [K/32][N][32] so a lane's B fragment is one contiguous 32B load.
// kbase lets us stack Wx and Wh into one combined K dimension.
// ---------------------------------------------------------------------------
__global__ void prep_weight(const float* __restrict__ src, _Float16* __restrict__ dst,
                            int K, int N, int kbase) {
  size_t i = (size_t)blockIdx.x * blockDim.x + threadIdx.x;
  size_t total = (size_t)K * N;
  if (i >= total) return;
  int k = (int)(i / N);
  int n = (int)(i - (size_t)k * N);
  int kk = kbase + k;
  dst[((size_t)(kk >> 5) * N + n) * 32 + (kk & 31)] = (_Float16)src[i];
}

// Prep kernel 2: zero recurrent state + barrier words (deterministic per call)
__global__ void prep_state(_Float16* h1a, _Float16* h1b, _Float16* h2a, _Float16* h2b,
                           float* c1, float* c2, unsigned* cnt, unsigned* gen) {
  size_t i = (size_t)blockIdx.x * blockDim.x + threadIdx.x;
  if (i < (size_t)B_ * H_) {
    _Float16 z = (_Float16)0.f;
    h1a[i] = z; h1b[i] = z; h2a[i] = z; h2b[i] = z;
    c1[i] = 0.f; c2[i] = 0.f;
  }
  if (i == 0) { *cnt = 0u; *gen = 0u; }
}

// ---------------------------------------------------------------------------
// Grid-wide barrier (persistent kernel, all NWG_ workgroups co-resident)
// ---------------------------------------------------------------------------
__device__ __forceinline__ void grid_sync(unsigned* cnt, unsigned* gen) {
  __threadfence();                 // release my writes to device scope
  __syncthreads();
  if (threadIdx.x == 0) {
    unsigned g = __hip_atomic_load(gen, __ATOMIC_RELAXED, __HIP_MEMORY_SCOPE_AGENT);
    unsigned prev = __hip_atomic_fetch_add(cnt, 1u, __ATOMIC_ACQ_REL, __HIP_MEMORY_SCOPE_AGENT);
    if (prev + 1u == (unsigned)NWG_) {
      __hip_atomic_store(cnt, 0u, __ATOMIC_RELAXED, __HIP_MEMORY_SCOPE_AGENT);
      __hip_atomic_fetch_add(gen, 1u, __ATOMIC_ACQ_REL, __HIP_MEMORY_SCOPE_AGENT);
    } else {
      while (__hip_atomic_load(gen, __ATOMIC_ACQUIRE, __HIP_MEMORY_SCOPE_AGENT) == g) {
        __builtin_amdgcn_s_sleep(2);
      }
    }
  }
  __syncthreads();
  __threadfence();                 // acquire: drop stale lines before reading peers' data
}

// ---------------------------------------------------------------------------
// Fragment loaders (wave32, contiguous k order used by this code in both A and B)
// B 32x16 f16: lanes 0-15 -> col, K 0..15 ; lanes 16-31 -> col, K 16..31.
// With the repacked [K/32][N][32] layout this is a single contiguous 32B load.
// ---------------------------------------------------------------------------
__device__ __forceinline__ v16h load_b_frag(const _Float16* __restrict__ Wt,
                                            int kchunk, int ncols, int col, int lane) {
  const _Float16* p = Wt + ((size_t)kchunk * ncols + col) * 32 + ((lane >> 4) << 4);
  return *(const v16h*)p;
}

// A 16x32 f16: lane(0-15)=row M, K 0..15; lane(16-31)=row M, K 16..31
__device__ __forceinline__ v16h load_a_f16(const _Float16* __restrict__ row, int kc, int koff) {
  const _Float16* p = row + kc + koff;
  v8h lo = *(const v8h*)(p);
  v8h hi = *(const v8h*)(p + 8);
  v16h r;
#pragma unroll
  for (int i = 0; i < 8; ++i) { r[i] = lo[i]; r[i + 8] = hi[i]; }
  return r;
}

__device__ __forceinline__ v16h load_a_f32(const float* __restrict__ row, int kc, int koff) {
  const float* p = row + kc + koff;
  v16h r;
#pragma unroll
  for (int i = 0; i < 8; ++i) {
    r[i]     = (_Float16)p[i];
    r[i + 8] = (_Float16)p[8 + i];
  }
  return r;
}

// ---------------------------------------------------------------------------
// Fused gate nonlinearity + cell update. Each wave drops its 4 accumulator
// tiles (its gate, its 16 cols, 64 rows) into LDS; then 2048 cells are updated
// elementwise (i,f,g,o all local to this workgroup by construction).
// C/D layout: lane<16 holds N=lane, M=v ; lane>=16 holds N=lane-16, M=v+8.
// ---------------------------------------------------------------------------
__device__ __forceinline__ void apply_gates(const v8f* acc, float* zbuf,
                                            const float* __restrict__ bias,
                                            float* __restrict__ cbuf,
                                            _Float16* __restrict__ hbuf,
                                            int m0, int j0, int gate, int jh,
                                            int lane, int tid) {
  int colr = jh + (lane & 15);
  int rofs = (lane >> 4) << 3;   // +8 rows for upper half-wave
#pragma unroll
  for (int m = 0; m < 4; ++m) {
#pragma unroll
    for (int v = 0; v < 8; ++v) {
      zbuf[((size_t)gate * 64 + m * 16 + rofs + v) * 32 + colr] = acc[m][v];
    }
  }
  __syncthreads();
#pragma unroll
  for (int u = 0; u < 8; ++u) {
    int cell = tid * 8 + u;                  // 0..2047 over 64 rows x 32 cols
    int mr = cell >> 5;
    int jj = cell & 31;
    int j  = j0 + jj;
    float zi = zbuf[(0 * 64 + mr) * 32 + jj] + bias[0 * H_ + j];
    float zf = zbuf[(1 * 64 + mr) * 32 + jj] + bias[1 * H_ + j];
    float zg = zbuf[(2 * 64 + mr) * 32 + jj] + bias[2 * H_ + j];
    float zo = zbuf[(3 * 64 + mr) * 32 + jj] + bias[3 * H_ + j];
    float ig = 1.f / (1.f + __expf(-zi));
    float fg = 1.f / (1.f + __expf(-zf));
    float gg = tanhf(zg);
    float og = 1.f / (1.f + __expf(-zo));
    size_t idx = (size_t)(m0 + mr) * H_ + j;
    float cn = fg * cbuf[idx] + ig * gg;
    cbuf[idx] = cn;
    hbuf[idx] = (_Float16)(og * tanhf(cn));
  }
}

// ---------------------------------------------------------------------------
// Persistent LSTM kernel: 64 WGs, 2 grid barriers per timestep.
// ---------------------------------------------------------------------------
__global__ void __launch_bounds__(256)
lstm_persistent(const float* __restrict__ inputs, const float* __restrict__ eps,
                const float* __restrict__ b1, const float* __restrict__ b2,
                const float* __restrict__ w_mean, const float* __restrict__ b_mean,
                const float* __restrict__ w_sigma, const float* __restrict__ b_sigma,
                const _Float16* __restrict__ W1t, const _Float16* __restrict__ W2t,
                _Float16* h1a, _Float16* h1b, _Float16* h2a, _Float16* h2b,
                float* c1, float* c2,
                unsigned* cnt, unsigned* gen, float* lossbuf, float* out) {
  __shared__ float zbuf[4 * 64 * 32];       // 32 KB

  const int wg   = blockIdx.x;              // 0..63
  const int mt   = wg & 3;
  const int nt   = wg >> 2;
  const int m0   = mt * 64;
  const int j0   = nt * 32;
  const int tid  = threadIdx.x;
  const int lane = tid & 31;
  const int w    = tid >> 5;                // wave 0..7
  const int gate = w >> 1;                  // i,f,g,o
  const int jh   = (w & 1) << 4;            // 0 or 16 within the 32-col tile
  const int colz = gate * H_ + j0 + jh + (lane & 15);  // column in 2048-wide gate matrix
  const int koff = ((lane >> 4) << 4);      // +16 K offset for upper half-wave (A frags)

  // Hoisted per-m row indices / offsets (invariant over t)
  int    rown[4];
  size_t hoff[4];
  const float* xbase[4];
#pragma unroll
  for (int m = 0; m < 4; ++m) {
    rown[m]  = m0 + m * 16 + (lane & 15);
    hoff[m]  = (size_t)rown[m] * H_;
    xbase[m] = inputs + (size_t)rown[m] * T_ * DIN_;
  }
  // Hoisted B-fragment base pointers (lane-resolved)
  const _Float16* W1p = W1t + (size_t)colz * 32 + ((lane >> 4) << 4);
  const _Float16* W2p = W2t + (size_t)colz * 32 + ((lane >> 4) << 4);

  const _Float16* h1r = h1a;  _Float16* h1w = h1b;
  const _Float16* h2r = h2a;  _Float16* h2w = h2b;

  for (int t = 0; t < T_; ++t) {
    v8f acc[4];

    // ---------------- layer 1: z1 = x_t @ Wx1 + h1 @ Wh1 ----------------
#pragma unroll
    for (int m = 0; m < 4; ++m) acc[m] = (v8f){};

    {
      const size_t xoff = (size_t)t * DIN_;
      // x-part: K = 0..255 (fp32 inputs, convert on the fly)
      for (int kc = 0; kc < DIN_; kc += 32) {
        v16h bf = *(const v16h*)(W1p + (size_t)(kc >> 5) * FOURH_ * 32);
        __builtin_prefetch(W1p + (size_t)((kc >> 5) + 1) * FOURH_ * 32, 0, 3);
#pragma unroll
        for (int m = 0; m < 4; ++m) {
          v16h af = load_a_f32(xbase[m] + xoff, kc, koff);
          acc[m] = __builtin_amdgcn_wmma_f32_16x16x32_f16(
              false, af, false, bf, (short)0, acc[m], false, false);
        }
      }
      // h-part: K = 256..767 (f16 recurrent state)
      const _Float16* W1h = W1p + (size_t)(DIN_ >> 5) * FOURH_ * 32;
      for (int kc = 0; kc < H_; kc += 32) {
        v16h bf = *(const v16h*)(W1h + (size_t)(kc >> 5) * FOURH_ * 32);
        __builtin_prefetch(W1h + (size_t)((kc >> 5) + 1) * FOURH_ * 32, 0, 3);
#pragma unroll
        for (int m = 0; m < 4; ++m) {
          v16h af = load_a_f16(h1r + hoff[m], kc, koff);
          acc[m] = __builtin_amdgcn_wmma_f32_16x16x32_f16(
              false, af, false, bf, (short)0, acc[m], false, false);
        }
      }
    }
    apply_gates(acc, zbuf, b1, c1, h1w, m0, j0, gate, jh, lane, tid);
    grid_sync(cnt, gen);   // all of h1w visible device-wide

    // ---------------- layer 2: z2 = h1 @ Wx2 + h2 @ Wh2 ----------------
#pragma unroll
    for (int m = 0; m < 4; ++m) acc[m] = (v8f){};

    {
      // h1-part: K = 0..511
      for (int kc = 0; kc < H_; kc += 32) {
        v16h bf = *(const v16h*)(W2p + (size_t)(kc >> 5) * FOURH_ * 32);
        __builtin_prefetch(W2p + (size_t)((kc >> 5) + 1) * FOURH_ * 32, 0, 3);
#pragma unroll
        for (int m = 0; m < 4; ++m) {
          v16h af = load_a_f16(h1w + hoff[m], kc, koff);
          acc[m] = __builtin_amdgcn_wmma_f32_16x16x32_f16(
              false, af, false, bf, (short)0, acc[m], false, false);
        }
      }
      // h2-part: K = 512..1023
      const _Float16* W2h = W2p + (size_t)(H_ >> 5) * FOURH_ * 32;
      for (int kc = 0; kc < H_; kc += 32) {
        v16h bf = *(const v16h*)(W2h + (size_t)(kc >> 5) * FOURH_ * 32);
        __builtin_prefetch(W2h + (size_t)((kc >> 5) + 1) * FOURH_ * 32, 0, 3);
#pragma unroll
        for (int m = 0; m < 4; ++m) {
          v16h af = load_a_f16(h2r + hoff[m], kc, koff);
          acc[m] = __builtin_amdgcn_wmma_f32_16x16x32_f16(
              false, af, false, bf, (short)0, acc[m], false, false);
        }
      }
    }
    apply_gates(acc, zbuf, b2, c2, h2w, m0, j0, gate, jh, lane, tid);
    grid_sync(cnt, gen);   // h2w / c2 visible; state complete for step t

    // double-buffer swap
    { const _Float16* tmp = h1r; h1r = h1w; h1w = (_Float16*)tmp; }
    { const _Float16* tmp = h2r; h2r = h2w; h2w = (_Float16*)tmp; }
  }

  // ---------------- latent projections + reparameterize + loss ----------------
  float lpart = 0.f;
#pragma unroll
  for (int r = 0; r < 2; ++r) {
    int idx = (wg * 256 + tid) * 2 + r;     // 0..32767 == (b, l)
    int b = idx >> 7;
    int l = idx & 127;
    const float* feat = c2 + (size_t)b * H_;   // reference uses last layer CELL state
    float mn = b_mean[l];
    float sg = b_sigma[l];
    for (int k = 0; k < H_; ++k) {
      float f = feat[k];
      mn += f * w_mean[(size_t)k * L_ + l];
      sg += f * w_sigma[(size_t)k * L_ + l];
    }
    float noise = 0.0f + 1.0f * eps[idx];   // NOISE_MEAN + NOISE_STD * eps
    out[idx] = mn + __expf(0.5f * sg) * noise;
    lpart += -0.5f * (1.f + sg - mn * mn - __expf(sg));
  }

  // deterministic loss reduction: LDS tree -> per-WG partial -> ordered sum
  float* red = zbuf;
  red[tid] = lpart;
  __syncthreads();
  for (int st = 128; st > 0; st >>= 1) {
    if (tid < st) red[tid] += red[tid + st];
    __syncthreads();
  }
  if (tid == 0) lossbuf[wg] = red[0];
  grid_sync(cnt, gen);
  if (wg == 0 && tid == 0) {
    float s = 0.f;
    for (int i = 0; i < NWG_; ++i) s += lossbuf[i];
    out[(size_t)B_ * L_] = s / (float)(B_ * L_);
  }
}

// ---------------------------------------------------------------------------
// Host launcher
// ---------------------------------------------------------------------------
extern "C" void kernel_launch(void* const* d_in, const int* in_sizes, int n_in,
                              void* d_out, int out_size, void* d_ws, size_t ws_size,
                              hipStream_t stream) {
  (void)in_sizes; (void)n_in; (void)out_size; (void)ws_size;

  const float* inputs  = (const float*)d_in[0];
  const float* eps     = (const float*)d_in[1];
  const float* Wx1     = (const float*)d_in[2];
  const float* Wh1     = (const float*)d_in[3];
  const float* b1      = (const float*)d_in[4];
  const float* Wx2     = (const float*)d_in[5];
  const float* Wh2     = (const float*)d_in[6];
  const float* b2      = (const float*)d_in[7];
  const float* w_mean  = (const float*)d_in[8];
  const float* b_mean  = (const float*)d_in[9];
  const float* w_sigma = (const float*)d_in[10];
  const float* b_sigma = (const float*)d_in[11];
  float* out = (float*)d_out;

  // Workspace layout (all chunks multiple of 256B)
  char* ws = (char*)d_ws;
  unsigned* cnt     = (unsigned*)(ws + 0);
  unsigned* gen     = (unsigned*)(ws + 4);
  float*    lossbuf = (float*)(ws + 256);
  size_t off = 512;
  const size_t hbytes = (size_t)B_ * H_ * sizeof(_Float16);   // 262144
  const size_t cbytes = (size_t)B_ * H_ * sizeof(float);      // 524288
  _Float16* h1a = (_Float16*)(ws + off); off += hbytes;
  _Float16* h1b = (_Float16*)(ws + off); off += hbytes;
  _Float16* h2a = (_Float16*)(ws + off); off += hbytes;
  _Float16* h2b = (_Float16*)(ws + off); off += hbytes;
  float*    c1  = (float*)(ws + off);    off += cbytes;
  float*    c2  = (float*)(ws + off);    off += cbytes;
  _Float16* W1t = (_Float16*)(ws + off); off += (size_t)K1_ * FOURH_ * sizeof(_Float16);
  _Float16* W2t = (_Float16*)(ws + off); off += (size_t)K2_ * FOURH_ * sizeof(_Float16);

  // Repack weights (f32 -> f16, fragment-friendly [K/32][N][32])
  {
    size_t n;
    n = (size_t)DIN_ * FOURH_;
    prep_weight<<<(unsigned)((n + 255) / 256), 256, 0, stream>>>(Wx1, W1t, DIN_, FOURH_, 0);
    n = (size_t)H_ * FOURH_;
    prep_weight<<<(unsigned)((n + 255) / 256), 256, 0, stream>>>(Wh1, W1t, H_, FOURH_, DIN_);
    prep_weight<<<(unsigned)((n + 255) / 256), 256, 0, stream>>>(Wx2, W2t, H_, FOURH_, 0);
    prep_weight<<<(unsigned)((n + 255) / 256), 256, 0, stream>>>(Wh2, W2t, H_, FOURH_, H_);
  }

  // Zero state + barrier words
  {
    size_t n = (size_t)B_ * H_;
    prep_state<<<(unsigned)((n + 255) / 256), 256, 0, stream>>>(
        h1a, h1b, h2a, h2b, c1, c2, cnt, gen);
  }

  // Persistent recurrent kernel
  lstm_persistent<<<NWG_, 256, 0, stream>>>(
      inputs, eps, b1, b2, w_mean, b_mean, w_sigma, b_sigma,
      W1t, W2t, h1a, h1b, h2a, h2b, c1, c2, cnt, gen, lossbuf, out);
}
